// PatchEmbedDCT_P_45775761441157
// MI455X (gfx1250) — matrix operand order repaired and checked by reading, not compiled, not run on hardware
//
#include <hip/hip_runtime.h>

// ---------------------------------------------------------------------------
// PatchEmbed + per-8x8-tile DCT, MI455X (gfx1250, wave32, WMMA).
//
// DCT folded into conv weights (w' = D^T w D per aligned 8x8 subtile) ->
// plain GEMM: [25088 x 768] = [25088 x 768] x [768 x 768]^T,
// A = raw x patches (stride==kernel, no im2col copy), B = w' (bf16).
// Compute-bound (AI ~192 FLOP/B) -> bf16 WMMA 16x16x32, f32 accumulate.
// x (77MB) stays L2-resident (192MB L2) across the 3 N-tile re-reads.
// Block tile 128x256, 8 waves, each wave 64x64 = 4x4 WMMA accumulators.
// Software pipeline: global->reg prefetch of stage k+1 overlaps the 16 WMMAs.
// ---------------------------------------------------------------------------

typedef __attribute__((ext_vector_type(16))) __bf16 v16bf;
typedef __attribute__((ext_vector_type(2)))  __bf16 v2bf;
typedef __attribute__((ext_vector_type(8)))  float  v8f;

__device__ __forceinline__ unsigned short f2bf(float f) {
    __bf16 h = (__bf16)f;                       // native v_cvt, RNE
    return __builtin_bit_cast(unsigned short, h);
}
__device__ __forceinline__ unsigned int pack2bf(float a, float b) {
    v2bf v;
    v.x = (__bf16)a;                            // -> v_cvt_pk_bf16_f32
    v.y = (__bf16)b;
    return __builtin_bit_cast(unsigned int, v);
}

// ---------------------------------------------------------------------------
// Kernel 1: fold 8x8 DCT into conv weights, emit bf16 Wt[768][768].
//   Wt[o][c*256 + (ti*8+k)*16 + (tj*8+l)] = sum_{u,v} D[u][k]*w8[u][v]*D[v][l]
// ---------------------------------------------------------------------------
__global__ __launch_bounds__(64)
void dct_fold_weights(const float* __restrict__ w, unsigned short* __restrict__ wt) {
    __shared__ float Dm[8][8];
    __shared__ float W8[8][8];
    const int t  = blockIdx.x;          // 0..9215
    const int o  = t / 12;
    const int r  = t - o * 12;
    const int c  = r >> 2;
    const int ti = (r >> 1) & 1;
    const int tj = r & 1;
    const int u  = threadIdx.x >> 3;
    const int v  = threadIdx.x & 7;

    const float s = (u == 0) ? 0.35355339059327373f : 0.5f;   // sqrt(1/8), sqrt(2/8)
    Dm[u][v] = s * cosf(3.14159265358979323846f * (2.0f * (float)v + 1.0f) * (float)u / 16.0f);
    W8[u][v] = w[((size_t)(o * 3 + c) * 16 + (ti * 8 + u)) * 16 + (tj * 8 + v)];
    __syncthreads();

    const int k = u, l = v;
    float acc = 0.0f;
    #pragma unroll
    for (int uu = 0; uu < 8; ++uu) {
        float row = 0.0f;
        #pragma unroll
        for (int vv = 0; vv < 8; ++vv) row += W8[uu][vv] * Dm[vv][l];
        acc += Dm[uu][k] * row;
    }
    wt[(size_t)o * 768 + c * 256 + (ti * 8 + k) * 16 + (tj * 8 + l)] = f2bf(acc);
}

// ---------------------------------------------------------------------------
// Kernel 2: GEMM with bf16 WMMA, software-pipelined global fetch.
//   C[m,n] = sum_k A[m,k] * Wt[n,k] + bias[n]
//   m = b*196 + py*14 + px ; k = c*256 + i*16 + j ; A[m,k] = x[b,c,16py+i,16px+j]
// ---------------------------------------------------------------------------
#define BM 128
#define BN 256
#define BK 32
#define LDST 40   // LDS row stride in bf16 elements (32 + 8 pad, keeps 16B align)

struct Frag32 { uint4 lo; uint4 hi; };   // 32 bytes -> v16bf

struct Stage {               // per-thread staged global data for one K-step
    float4 f0, f1, f2, f3;   // 16 floats of an x image row segment
    uint4  d0, d1, d2, d3;   // 32 bf16 of a Wt row
};

__device__ __forceinline__ void load_stage(Stage& s, int kt, int lh,
                                           const float* __restrict__ xbase,
                                           const unsigned short* __restrict__ wrow) {
    const int c    = kt >> 8;            // channel
    const int i8   = (kt >> 4) & 15;     // even patch row
    const float* src = xbase + (size_t)c * 224 * 224 + (size_t)(i8 + lh) * 224;
    s.f0 = *(const float4*)(src + 0);
    s.f1 = *(const float4*)(src + 4);
    s.f2 = *(const float4*)(src + 8);
    s.f3 = *(const float4*)(src + 12);
    const uint4* bsrc = (const uint4*)(wrow + kt);
    s.d0 = bsrc[0];
    s.d1 = bsrc[1];
    s.d2 = bsrc[2];
    s.d3 = bsrc[3];
}

__global__ __launch_bounds__(256)
void dct_patch_gemm(const float* __restrict__ x,
                    const unsigned short* __restrict__ wt,
                    const float* __restrict__ bias,
                    float* __restrict__ out) {
    __shared__ unsigned short Asb[BM * LDST];   // A tile bf16, [row][k]
    __shared__ unsigned short Bsb[BN * LDST];   // B tile bf16, [n][k]

    const int tid = threadIdx.x;
    const int m0  = blockIdx.x * BM;   // 196 tiles
    const int n0  = blockIdx.y * BN;   // 3 tiles

    // ---- A staging assignment: row = tid/2, half = tid&1 (16 of 32 K) ----
    const int lr = tid >> 1;           // 0..127
    const int lh = tid & 1;

    // A-row geometry (constant over K loop)
    const int m    = m0 + lr;
    const int bimg = m / 196;
    const int p    = m - bimg * 196;
    const int py   = p / 14;
    const int px   = p - py * 14;
    const float* xbase = x + (size_t)bimg * 3 * 224 * 224
                           + (size_t)(py * 16) * 224 + px * 16;
    const unsigned short* wrow = wt + (size_t)(n0 + tid) * 768;  // B: thread owns row

    // ---- wave decomposition ----
    const int wave    = tid >> 5;
    const int lane    = tid & 31;
    const int wm      = wave & 1;      // 2 waves over M (64 rows each)
    const int wn      = wave >> 1;     // 4 waves over N (64 cols each)
    const int lane_r  = lane & 15;
    const int lane_hi = lane >> 4;

    v8f acc[4][4];
    #pragma unroll
    for (int i = 0; i < 4; ++i)
        #pragma unroll
        for (int j = 0; j < 4; ++j)
            acc[i][j] = (v8f){0.f,0.f,0.f,0.f,0.f,0.f,0.f,0.f};

    Stage st;
    load_stage(st, 0, lh, xbase, wrow);          // prologue fetch

    for (int kt = 0; kt < 768; kt += BK) {
        // ---- commit staged regs to LDS (A: f32 -> bf16 native cvt) ----
        {
            unsigned short* dst = &Asb[lr * LDST + lh * 16];
            ((uint2*)dst)[0] = make_uint2(pack2bf(st.f0.x, st.f0.y), pack2bf(st.f0.z, st.f0.w));
            ((uint2*)dst)[1] = make_uint2(pack2bf(st.f1.x, st.f1.y), pack2bf(st.f1.z, st.f1.w));
            ((uint2*)dst)[2] = make_uint2(pack2bf(st.f2.x, st.f2.y), pack2bf(st.f2.z, st.f2.w));
            ((uint2*)dst)[3] = make_uint2(pack2bf(st.f3.x, st.f3.y), pack2bf(st.f3.z, st.f3.w));
            unsigned short* bdst = &Bsb[tid * LDST];
            ((uint4*)bdst)[0] = st.d0;
            ((uint4*)bdst)[1] = st.d1;
            ((uint4*)bdst)[2] = st.d2;
            ((uint4*)bdst)[3] = st.d3;
        }
        __syncthreads();

        // ---- prefetch next stage: VMEM latency overlaps the WMMAs below ----
        const int ktn = kt + BK;
        if (ktn < 768)                             // uniform branch (EXEC stays full)
            load_stage(st, ktn, lh, xbase, wrow);

        // ---- fragments per ISA VGPR layouts ----
        // A 16x32 bf16: lanes 0-15 -> K 0..7 & 16..23 ; lanes 16-31 -> K 8..15 & 24..31
        v16bf afrag[4];
        #pragma unroll
        for (int ms = 0; ms < 4; ++ms) {
            const unsigned short* ap =
                &Asb[(wm * 64 + ms * 16 + lane_r) * LDST + lane_hi * 8];
            Frag32 f;
            f.lo = *(const uint4*)(ap);        // K = lane_hi*8 + 0..7
            f.hi = *(const uint4*)(ap + 16);   // K = 16 + lane_hi*8 + 0..7
            afrag[ms] = __builtin_bit_cast(v16bf, f);
        }
        // B 32x16 bf16: lanes 0-15 -> K 0..7 & 16..23 of col n ; lanes 16-31 -> K 8..15 & 24..31
        v16bf bfrag[4];
        #pragma unroll
        for (int ns = 0; ns < 4; ++ns) {
            const unsigned short* bp =
                &Bsb[(wn * 64 + ns * 16 + lane_r) * LDST + lane_hi * 8];
            Frag32 f;
            f.lo = *(const uint4*)(bp);
            f.hi = *(const uint4*)(bp + 16);
            bfrag[ns] = __builtin_bit_cast(v16bf, f);
        }

        #pragma unroll
        for (int ms = 0; ms < 4; ++ms)
            #pragma unroll
            for (int ns = 0; ns < 4; ++ns)
                acc[ms][ns] = __builtin_amdgcn_wmma_f32_16x16x32_bf16(
                    /*neg_a=*/false, afrag[ms],
                    /*neg_b=*/false, bfrag[ns],
                    /*c_mod=*/(short)0, acc[ms][ns],
                    /*reuse_a=*/false, /*reuse_b=*/false);

        __syncthreads();
    }

    // ---- epilogue: C/D layout -> M = vgpr + 8*lane_hi, N = lane_r ; +bias ----
    #pragma unroll
    for (int ns = 0; ns < 4; ++ns) {
        const int col = n0 + wn * 64 + ns * 16 + lane_r;
        const float bv = bias[col];
        #pragma unroll
        for (int ms = 0; ms < 4; ++ms) {
            const int row0 = m0 + wm * 64 + ms * 16 + lane_hi * 8;
            #pragma unroll
            for (int r = 0; r < 8; ++r)
                out[(size_t)(row0 + r) * 768 + col] = acc[ms][ns][r] + bv;
        }
    }
}

// ---------------------------------------------------------------------------
extern "C" void kernel_launch(void* const* d_in, const int* in_sizes, int n_in,
                              void* d_out, int out_size, void* d_ws, size_t ws_size,
                              hipStream_t stream) {
    const float* x = (const float*)d_in[0];   // [128,3,224,224]
    const float* w = (const float*)d_in[1];   // [768,3,16,16]
    const float* b = (const float*)d_in[2];   // [768]
    float* out = (float*)d_out;               // [128,196,768]
    unsigned short* wtb = (unsigned short*)d_ws;  // bf16 Wt[768][768] (1.18 MB)

    dct_fold_weights<<<dim3(9216), dim3(64), 0, stream>>>(w, wtb);

    dim3 grid(196, 3);   // 25088/128 M-tiles x 768/256 N-tiles
    dct_patch_gemm<<<grid, dim3(256), 0, stream>>>(x, wtb, b, out);
}
